// MambaBlock_62448824483847
// MI455X (gfx1250) — matrix operand order repaired and checked by reading, not compiled, not run on hardware
//
#include <hip/hip_runtime.h>
#include <hip/hip_bf16.h>
#include <stdint.h>

// Problem constants (match reference)
#define LSEQ    2048
#define DMODEL  1024
#define DSTATE  16
#define DINNER  2048
#define NPROJ   4096      // 2*D_INNER
#define XPS     48        // padded row stride for xp (33 valid)
#define LCH     128       // scan chunk length
#define NCH     16        // number of scan chunks (LSEQ/LCH)

typedef __attribute__((ext_vector_type(16))) __bf16       v16bf;
typedef __attribute__((ext_vector_type(8)))  float        v8f;
typedef __attribute__((ext_vector_type(4)))  unsigned int u32x4;
typedef int gccv4i __attribute__((vector_size(16)));   // matches async-LDS builtin param

union Frag { v16bf v; u32x4 q[2]; };
struct FragSet { Frag a0, a1, b0, b1, b2, b3; };

__device__ __forceinline__ unsigned short f2bf(float f) {
  unsigned int u = __float_as_uint(f);
  u += 0x7FFFu + ((u >> 16) & 1u);       // round-to-nearest-even
  return (unsigned short)(u >> 16);
}
__device__ __forceinline__ float softplusf(float x) {
  return (x > 20.0f) ? x : log1pf(__expf(x));
}
__device__ __forceinline__ float siluf(float x) {
  return x / (1.0f + __expf(-x));
}

// ---------------- fp32 -> bf16 conversion (packed stores) ----------------
__global__ void k_cvt_bf16(const float* __restrict__ s, unsigned short* __restrict__ d, int n) {
  int i = (blockIdx.x * blockDim.x + threadIdx.x) * 2;
  if (i + 1 < n) {
    unsigned int lo = f2bf(s[i]);
    unsigned int hi = f2bf(s[i + 1]);
    *(unsigned int*)(d + i) = lo | (hi << 16);
  }
}

// ---------------- bf16 WMMA GEMM: C[M,N] = A[M,K] * W[N,K]^T ----------------
// Block = 128 threads = 4 waves; each wave computes a 32x64 tile (2 A-frags x
// 4 B-frags -> 8 WMMAs per stage) with ping-pong double buffering so global
// loads overlap the WMMA pipe. launch_bounds(128,1) lifts the VGPR cap so the
// ~190-VGPR working set stays in registers (no scratch spills).
// grid.x = M/32, grid.y = N/256. K % 64 == 0.
__global__ __launch_bounds__(128, 1)
void k_gemm_bf16(const unsigned short* __restrict__ A,
                 const unsigned short* __restrict__ W,
                 float* __restrict__ C, int N, int K) {
  const int lane = threadIdx.x & 31;
  const int wave = threadIdx.x >> 5;
  const int r    = lane & 15;        // row-within-tile / output column
  const int hi   = lane >> 4;        // half-wave selector
  const int m0   = blockIdx.x << 5;
  const int n0   = (blockIdx.y << 8) + (wave << 6);

  const unsigned short* a0p = A + (size_t)(m0 +  0 + r) * K;
  const unsigned short* a1p = A + (size_t)(m0 + 16 + r) * K;
  const unsigned short* w0  = W + (size_t)(n0 +  0 + r) * K;
  const unsigned short* w1  = W + (size_t)(n0 + 16 + r) * K;
  const unsigned short* w2  = W + (size_t)(n0 + 32 + r) * K;
  const unsigned short* w3  = W + (size_t)(n0 + 48 + r) * K;

  v8f c[8] = {};

  // A 16x32 bf16 frag: lane(r,hi) holds 16 contiguous halfs from k+hi*16 of A row m=r
  // B 32x16 bf16 frag: lane(r,hi) holds 16 contiguous halfs from k+hi*16 of W row n=r
  // pf=true additionally issues prefetches; the W ones share the load address registers
  // (constant +512B immediate offset -> global_prefetch_b8 with folded offset).
  auto LOAD = [&](FragSet& F, int k, bool pf) {
    F.a0.q[0] = *(const u32x4*)(a0p + k + hi * 16);
    F.a0.q[1] = *(const u32x4*)(a0p + k + hi * 16 + 8);
    F.a1.q[0] = *(const u32x4*)(a1p + k + hi * 16);
    F.a1.q[1] = *(const u32x4*)(a1p + k + hi * 16 + 8);
    const int kb = k + hi * 16;
    F.b0.q[0] = *(const u32x4*)(w0 + kb); F.b0.q[1] = *(const u32x4*)(w0 + kb + 8);
    F.b1.q[0] = *(const u32x4*)(w1 + kb); F.b1.q[1] = *(const u32x4*)(w1 + kb + 8);
    F.b2.q[0] = *(const u32x4*)(w2 + kb); F.b2.q[1] = *(const u32x4*)(w2 + kb + 8);
    F.b3.q[0] = *(const u32x4*)(w3 + kb); F.b3.q[1] = *(const u32x4*)(w3 + kb + 8);
    if (pf) {   // speculative stream-ahead; OOB prefetch is silently dropped
      __builtin_prefetch(a0p + k + hi * 8 + 256, 0, 3);
      __builtin_prefetch(a1p + k + hi * 8 + 256, 0, 3);
      __builtin_prefetch(w0 + kb + 256, 0, 3);
      __builtin_prefetch(w1 + kb + 256, 0, 3);
      __builtin_prefetch(w2 + kb + 256, 0, 3);
      __builtin_prefetch(w3 + kb + 256, 0, 3);
    }
  };
  auto MMA = [&](FragSet& F) {
    c[0] = __builtin_amdgcn_wmma_f32_16x16x32_bf16(false, F.a0.v, false, F.b0.v, (short)0, c[0], false, false);
    c[1] = __builtin_amdgcn_wmma_f32_16x16x32_bf16(false, F.a0.v, false, F.b1.v, (short)0, c[1], false, false);
    c[2] = __builtin_amdgcn_wmma_f32_16x16x32_bf16(false, F.a0.v, false, F.b2.v, (short)0, c[2], false, false);
    c[3] = __builtin_amdgcn_wmma_f32_16x16x32_bf16(false, F.a0.v, false, F.b3.v, (short)0, c[3], false, false);
    c[4] = __builtin_amdgcn_wmma_f32_16x16x32_bf16(false, F.a1.v, false, F.b0.v, (short)0, c[4], false, false);
    c[5] = __builtin_amdgcn_wmma_f32_16x16x32_bf16(false, F.a1.v, false, F.b1.v, (short)0, c[5], false, false);
    c[6] = __builtin_amdgcn_wmma_f32_16x16x32_bf16(false, F.a1.v, false, F.b2.v, (short)0, c[6], false, false);
    c[7] = __builtin_amdgcn_wmma_f32_16x16x32_bf16(false, F.a1.v, false, F.b3.v, (short)0, c[7], false, false);
  };

  FragSet cur, nxt;
  LOAD(cur, 0, false);
  for (int k = 0; k < K; k += 64) {
    LOAD(nxt, k + 32, true);                    // overlap with MMA(cur)
    MMA(cur);
    if (k + 64 < K) LOAD(cur, k + 64, false);   // overlap with MMA(nxt)
    MMA(nxt);
  }

  // C/D layout: VGPR v, lanes 0-15 -> (M=v, N=lane); lanes 16-31 -> (M=v+8, N=lane-16)
#pragma unroll
  for (int v = 0; v < 8; ++v) {
    size_t r0 = (size_t)(m0 + v + hi * 8) * N + n0 + r;
    size_t r1 = (size_t)(m0 + 16 + v + hi * 8) * N + n0 + r;
    C[r0]      = c[0][v];
    C[r0 + 16] = c[1][v];
    C[r0 + 32] = c[2][v];
    C[r0 + 48] = c[3][v];
    C[r1]      = c[4][v];
    C[r1 + 16] = c[5][v];
    C[r1 + 32] = c[6][v];
    C[r1 + 48] = c[7][v];
  }
}

// ---------------- depthwise causal conv (K=4) + bias + SiLU ----------------
__global__ void k_conv_silu(const float* __restrict__ xz, const float* __restrict__ cw,
                            const float* __restrict__ cb, float* __restrict__ xc) {
  const int idx = blockIdx.x * blockDim.x + threadIdx.x;   // over LSEQ*DINNER
  const int d = idx & (DINNER - 1);
  const int l = idx >> 11;
  float acc = cb[d];
#pragma unroll
  for (int t = 0; t < 4; ++t) {
    const int ll = l - 3 + t;
    if (ll >= 0) acc = fmaf(xz[(size_t)ll * NPROJ + d], cw[d * 4 + t], acc);
  }
  xc[idx] = siluf(acc);
}

// ---------------- x_proj: xp[l, 0..32] = xc[l,:] . x_proj_w[j,:] ----------------
__global__ void k_xproj(const float* __restrict__ xc, const float* __restrict__ xpw,
                        float* __restrict__ xp) {
  __shared__ float row[DINNER];
  __shared__ float red[256];
  const int l = blockIdx.x;
  const int t = threadIdx.x;
#if __has_builtin(__builtin_amdgcn_global_load_async_to_lds_b128)
  // Stage the 8KB activation row in LDS with the CDNA5 async copy path
  // (ASYNCcnt), 16B per lane per sweep.
  for (int off = t * 4; off < DINNER; off += 1024) {
    gccv4i __attribute__((address_space(1)))* gp =
        (gccv4i __attribute__((address_space(1)))*)(xc + (size_t)l * DINNER + off);
    gccv4i __attribute__((address_space(3)))* lp =
        (gccv4i __attribute__((address_space(3)))*)&row[off];
    __builtin_amdgcn_global_load_async_to_lds_b128(gp, lp, 0, 0);
  }
#if __has_builtin(__builtin_amdgcn_s_wait_asynccnt)
  __builtin_amdgcn_s_wait_asynccnt(0);
#else
  asm volatile("s_wait_asynccnt 0" ::: "memory");
#endif
  __syncthreads();
#else
  for (int d = t; d < DINNER; d += 256) row[d] = xc[(size_t)l * DINNER + d];
  __syncthreads();
#endif
  for (int j = 0; j < 33; ++j) {
    const float* w = xpw + (size_t)j * DINNER;
    float acc = 0.f;
    for (int d = t; d < DINNER; d += 256) acc = fmaf(row[d], w[d], acc);
    red[t] = acc;
    __syncthreads();
    for (int s2 = 128; s2 > 0; s2 >>= 1) {
      if (t < s2) red[t] += red[t + s2];
      __syncthreads();
    }
    if (t == 0) xp[(size_t)l * XPS + j] = red[0];
    __syncthreads();
  }
}

// ---------------- scan phase 1: chunk-local scans (h0 = 0) ----------------
// grid = (DINNER/32, NCH), block = 32 (one wave, one channel per lane)
__global__ void k_scan_phase1(const float* __restrict__ xp, const float* __restrict__ xc,
                              const float* __restrict__ Alog, const float* __restrict__ dtw,
                              const float* __restrict__ dtb,
                              float* __restrict__ ys, float* __restrict__ Hloc,
                              float* __restrict__ dtsum) {
  const int d = blockIdx.x * 32 + threadIdx.x;
  const int c = blockIdx.y;
  float An[DSTATE], h[DSTATE];
#pragma unroll
  for (int n = 0; n < DSTATE; ++n) { An[n] = -__expf(Alog[d * DSTATE + n]); h[n] = 0.f; }
  const float wd = dtw[d], bd = dtb[d];
  float sdt = 0.f;
  const int l0 = c * LCH;
  for (int l = l0; l < l0 + LCH; ++l) {
    const float* xpl = xp + (size_t)l * XPS;          // uniform across lanes
    const float dt = softplusf(fmaf(xpl[0], wd, bd));
    sdt += dt;
    const float dxc = dt * xc[(size_t)l * DINNER + d];
    float y = 0.f;
#pragma unroll
    for (int n = 0; n < DSTATE; ++n) {
      h[n] = fmaf(__expf(dt * An[n]), h[n], dxc * xpl[1 + n]);  // B[l,n]
      y = fmaf(h[n], xpl[17 + n], y);                            // C[l,n]
    }
    ys[(size_t)l * DINNER + d] = y;
  }
#pragma unroll
  for (int n = 0; n < DSTATE; ++n) Hloc[((size_t)c * DINNER + d) * DSTATE + n] = h[n];
  dtsum[c * DINNER + d] = sdt;
}

// ---------------- scan combine: propagate chunk-entry states ----------------
// cumulative chunk decay P_c = exp(A_n * sum(dt over chunk))
__global__ void k_scan_combine(const float* __restrict__ Hloc, const float* __restrict__ dtsum,
                               const float* __restrict__ Alog, float* __restrict__ H0) {
  const int idx = blockIdx.x * blockDim.x + threadIdx.x;  // d*16+n over DINNER*DSTATE
  const int d = idx >> 4;
  const float An = -__expf(Alog[idx]);
  float h = 0.f;
  for (int c = 0; c < NCH; ++c) {
    H0[(size_t)c * DINNER * DSTATE + idx] = h;
    h = fmaf(__expf(An * dtsum[c * DINNER + d]), h, Hloc[(size_t)c * DINNER * DSTATE + idx]);
  }
}

// ---------------- scan phase 2: add h0 * cumprod(dA) . C correction ----------------
// grid = (DINNER/32, NCH-1); chunk 0 needs no correction.
__global__ void k_scan_phase2(const float* __restrict__ xp, const float* __restrict__ Alog,
                              const float* __restrict__ dtw, const float* __restrict__ dtb,
                              const float* __restrict__ H0, float* __restrict__ ys) {
  const int d = blockIdx.x * 32 + threadIdx.x;
  const int c = blockIdx.y + 1;
  float An[DSTATE], q[DSTATE];
#pragma unroll
  for (int n = 0; n < DSTATE; ++n) {
    An[n] = -__expf(Alog[d * DSTATE + n]);
    q[n] = H0[((size_t)c * DINNER + d) * DSTATE + n];
  }
  const float wd = dtw[d], bd = dtb[d];
  const int l0 = c * LCH;
  for (int l = l0; l < l0 + LCH; ++l) {
    const float* xpl = xp + (size_t)l * XPS;
    const float dt = softplusf(fmaf(xpl[0], wd, bd));
    float y = 0.f;
#pragma unroll
    for (int n = 0; n < DSTATE; ++n) {
      q[n] *= __expf(dt * An[n]);
      y = fmaf(q[n], xpl[17 + n], y);
    }
    ys[(size_t)l * DINNER + d] += y;
  }
}

// ---------------- y = ys + xc*D; ybf16 = bf16(y * silu(z)) ----------------
__global__ void k_ygate(const float* __restrict__ ys, const float* __restrict__ xc,
                        const float* __restrict__ Dw, const float* __restrict__ xz,
                        unsigned short* __restrict__ ybf) {
  const int idx = blockIdx.x * blockDim.x + threadIdx.x;
  const int d = idx & (DINNER - 1);
  const int l = idx >> 11;
  const float y = fmaf(xc[idx], Dw[d], ys[idx]);
  const float z = xz[(size_t)l * NPROJ + DINNER + d];
  ybf[idx] = f2bf(y * siluf(z));
}

// ---------------- launcher ----------------
extern "C" void kernel_launch(void* const* d_in, const int* in_sizes, int n_in,
                              void* d_out, int out_size, void* d_ws, size_t ws_size,
                              hipStream_t stream) {
  const float* x    = (const float*)d_in[0];
  const float* wi   = (const float*)d_in[1];   // (4096,1024)
  const float* cw   = (const float*)d_in[2];   // (2048,1,4)
  const float* cb   = (const float*)d_in[3];
  const float* xpw  = (const float*)d_in[4];   // (33,2048)
  const float* dtw  = (const float*)d_in[5];   // (2048,1)
  const float* dtb  = (const float*)d_in[6];
  const float* Alog = (const float*)d_in[7];   // (2048,16)
  const float* Dw   = (const float*)d_in[8];
  const float* wo   = (const float*)d_in[9];   // (1024,2048)
  float* out = (float*)d_out;

  char* p = (char*)d_ws;
  auto alloc = [&](size_t bytes) { char* q = p; p += (bytes + 255) & ~(size_t)255; return q; };
  unsigned short* xbf = (unsigned short*)alloc((size_t)LSEQ * DMODEL * 2);
  unsigned short* wib = (unsigned short*)alloc((size_t)NPROJ * DMODEL * 2);
  unsigned short* wob = (unsigned short*)alloc((size_t)DMODEL * DINNER * 2);
  float* xz  = (float*)alloc((size_t)LSEQ * NPROJ * 4);
  float* xc  = (float*)alloc((size_t)LSEQ * DINNER * 4);
  float* xp  = (float*)alloc((size_t)LSEQ * XPS * 4);
  float* ys  = (float*)alloc((size_t)LSEQ * DINNER * 4);
  unsigned short* ybf = (unsigned short*)alloc((size_t)LSEQ * DINNER * 2);
  float* Hloc = (float*)alloc((size_t)NCH * DINNER * DSTATE * 4);
  float* H0   = (float*)alloc((size_t)NCH * DINNER * DSTATE * 4);
  float* dts  = (float*)alloc((size_t)NCH * DINNER * 4);

  // 1) convert x, in_proj_w, out_proj_w to bf16
  k_cvt_bf16<<<(LSEQ * DMODEL) / 512, 256, 0, stream>>>(x, xbf, LSEQ * DMODEL);
  k_cvt_bf16<<<(NPROJ * DMODEL) / 512, 256, 0, stream>>>(wi, wib, NPROJ * DMODEL);
  k_cvt_bf16<<<(DMODEL * DINNER) / 512, 256, 0, stream>>>(wo, wob, DMODEL * DINNER);

  // 2) xz = x @ in_proj_w^T   (M=2048, N=4096, K=1024)
  k_gemm_bf16<<<dim3(LSEQ / 32, NPROJ / 256), 128, 0, stream>>>(xbf, wib, xz, NPROJ, DMODEL);

  // 3) depthwise causal conv + SiLU
  k_conv_silu<<<(LSEQ * DINNER) / 256, 256, 0, stream>>>(xz, cw, cb, xc);

  // 4) xp = xc @ x_proj_w^T (N=33)
  k_xproj<<<LSEQ, 256, 0, stream>>>(xc, xpw, xp);

  // 5) chunked selective scan
  k_scan_phase1<<<dim3(DINNER / 32, NCH), 32, 0, stream>>>(xp, xc, Alog, dtw, dtb, ys, Hloc, dts);
  k_scan_combine<<<(DINNER * DSTATE) / 256, 256, 0, stream>>>(Hloc, dts, Alog, H0);
  k_scan_phase2<<<dim3(DINNER / 32, NCH - 1), 32, 0, stream>>>(xp, Alog, dtw, dtb, H0, ys);

  // 6) gate: ybf = bf16((ys + xc*D) * silu(z))
  k_ygate<<<(LSEQ * DINNER) / 256, 256, 0, stream>>>(ys, xc, Dw, xz, ybf);

  // 7) out = y_gated @ out_proj_w^T  (M=2048, N=1024, K=2048)
  k_gemm_bf16<<<dim3(LSEQ / 32, DMODEL / 256), 128, 0, stream>>>(ybf, wob, out, DMODEL, DINNER);
}
